// RCU_89610197664397
// MI455X (gfx1250) — compile-verified, matrix operand order checked by reading, not run on hardware
//
#include <hip/hip_runtime.h>
#include <math.h>

// ---------------------------------------------------------------------------
// Dual-attention block for MI455X (gfx1250, wave32, WMMA).
// Heavy path: flash-style two-pass spatial attention using
// v_wmma_f32_16x16x32_f16, chaining score-WMMA D fragments directly into the
// B operand of the output WMMA (no materialized 4096x4096 matrix).
// ---------------------------------------------------------------------------

typedef __attribute__((ext_vector_type(16))) _Float16 v16h;
typedef __attribute__((ext_vector_type(8)))  float    v8f;

#define B_    8
#define CH_   32
#define HW_   4096
#define NT_   256      // 16-wide tiles along the 4096 axis
#define NPAIR_ 128     // 32-wide n-tile pairs (K of second WMMA)

static __device__ __forceinline__ v8f wmma_f16(v16h a, v16h b, v8f c) {
    return __builtin_amdgcn_wmma_f32_16x16x32_f16(
        /*neg_a=*/false, a, /*neg_b=*/false, b,
        /*c_mod=*/(short)0, c, /*reuse_a=*/false, /*reuse_b=*/false);
}

// ---------------------------------------------------------------------------
// K1: x1_co = relu(conv3x3(interleave(x groups, y), conv_w) + conv_b)
// cat channel ci: group g=ci/5, r=ci%5; r<4 -> x[g*4+r], r==4 -> y
// ---------------------------------------------------------------------------
__global__ void k_conv1(const float* __restrict__ x, const float* __restrict__ y,
                        const float* __restrict__ w, const float* __restrict__ bias,
                        float* __restrict__ out) {
    int idx = blockIdx.x * blockDim.x + threadIdx.x;
    if (idx >= B_ * CH_ * HW_) return;
    int n = idx & 4095, oc = (idx >> 12) & 31, b = idx >> 17;
    int h = n >> 6, wq = n & 63;
    float acc = bias[oc];
    for (int ci = 0; ci < 40; ++ci) {
        int g = ci / 5, r = ci % 5;
        const float* src = (r == 4) ? (y + (size_t)b * HW_)
                                    : (x + ((size_t)(b * CH_ + g * 4 + r)) * HW_);
        const float* wp = w + ((size_t)(oc * 40 + ci)) * 9;
        #pragma unroll
        for (int kh = 0; kh < 3; ++kh) {
            int hh = h + kh - 1;
            if (hh < 0 || hh > 63) continue;
            #pragma unroll
            for (int kw = 0; kw < 3; ++kw) {
                int ww = wq + kw - 1;
                if (ww < 0 || ww > 63) continue;
                acc += wp[kh * 3 + kw] * src[hh * 64 + ww];
            }
        }
    }
    out[idx] = fmaxf(acc, 0.f);
}

// ---------------------------------------------------------------------------
// K2: x2_co = conv3x3(y, c132_w) + c132_b
// ---------------------------------------------------------------------------
__global__ void k_conv2(const float* __restrict__ y, const float* __restrict__ w,
                        const float* __restrict__ bias, float* __restrict__ out) {
    int idx = blockIdx.x * blockDim.x + threadIdx.x;
    if (idx >= B_ * CH_ * HW_) return;
    int n = idx & 4095, oc = (idx >> 12) & 31, b = idx >> 17;
    int h = n >> 6, wq = n & 63;
    const float* src = y + (size_t)b * HW_;
    const float* wp = w + (size_t)oc * 9;
    float acc = bias[oc];
    #pragma unroll
    for (int kh = 0; kh < 3; ++kh) {
        int hh = h + kh - 1;
        if (hh < 0 || hh > 63) continue;
        #pragma unroll
        for (int kw = 0; kw < 3; ++kw) {
            int ww = wq + kw - 1;
            if (ww < 0 || ww > 63) continue;
            acc += wp[kh * 3 + kw] * src[hh * 64 + ww];
        }
    }
    out[idx] = acc;
}

// ---------------------------------------------------------------------------
// K3: six 1x1 projections. q1,k2 packed straight into WMMA A/B f16 fragment
// layout (per-lane 16 halfs); yq packed into the A layout used by pass 2
// (M=c, K=n). k1,q2,yk stay f32 (channel attention is tiny).
// Fragment layout convention (16-bit, matches ISA table):
//   lane l<16 : row/col l  holds K {0..7, 16..23}
//   lane l>=16: row/col l-16 holds K {8..15, 24..31}
// ---------------------------------------------------------------------------
__global__ void k_proj(const float* __restrict__ x1_co, const float* __restrict__ x2_co,
                       const float* __restrict__ q1w, const float* __restrict__ q1b,
                       const float* __restrict__ k1w, const float* __restrict__ k1b,
                       const float* __restrict__ q2w, const float* __restrict__ q2b,
                       const float* __restrict__ k2w, const float* __restrict__ k2b,
                       const float* __restrict__ q3w, const float* __restrict__ q3b,
                       const float* __restrict__ k3w, const float* __restrict__ k3b,
                       _Float16* __restrict__ q1p, _Float16* __restrict__ k2p,
                       _Float16* __restrict__ yqp,
                       float* __restrict__ k1o, float* __restrict__ q2o,
                       float* __restrict__ yko) {
    int idx = blockIdx.x * blockDim.x + threadIdx.x;
    if (idx >= B_ * HW_) return;
    int b = idx >> 12, n = idx & 4095;
    float v1[32], v2[32];
    #pragma unroll
    for (int c = 0; c < 32; ++c) {
        v1[c] = x1_co[((size_t)(b * 32 + c)) * HW_ + n];
        v2[c] = x2_co[((size_t)(b * 32 + c)) * HW_ + n];
    }
    int tile = n >> 4, l0n = n & 15;
    int kk = n & 31, np = n >> 5;
    int hik = (kk >> 3) & 1;
    int halfk = (kk & 7) + (kk >= 16 ? 8 : 0);
    for (int oc = 0; oc < 32; ++oc) {
        float a1 = q1b[oc], a2 = k1b[oc], a3 = q2b[oc];
        float a4 = k2b[oc], a5 = q3b[oc], a6 = k3b[oc];
        #pragma unroll
        for (int i = 0; i < 32; ++i) {
            a1 += q1w[oc * 32 + i] * v1[i];
            a2 += k1w[oc * 32 + i] * v1[i];
            a3 += q2w[oc * 32 + i] * v1[i];
            a4 += k2w[oc * 32 + i] * v1[i];
            a5 += q3w[oc * 32 + i] * v2[i];
            a6 += k3w[oc * 32 + i] * v2[i];
        }
        // q1/k2 fragment scatter keyed by (K=oc, col/row=n)
        int hi = (oc >> 3) & 1;
        int half = (oc & 7) + (oc >= 16 ? 8 : 0);
        int lane = l0n + hi * 16;
        size_t dst = (((size_t)(b * NT_ + tile)) * 32 + lane) * 16 + half;
        q1p[dst] = (_Float16)a1;
        k2p[dst] = (_Float16)a4;
        // yq A-layout scatter keyed by (M=oc, K=n)
        int ct = oc >> 4, l0c = oc & 15;
        int lanek = l0c + hik * 16;
        size_t dsty = ((((size_t)(b * 2 + ct)) * NPAIR_ + np) * 32 + lanek) * 16 + halfk;
        yqp[dsty] = (_Float16)a5;
        k1o[((size_t)(b * 32 + oc)) * HW_ + n] = a2;
        q2o[((size_t)(b * 32 + oc)) * HW_ + n] = a3;
        yko[((size_t)(b * 32 + oc)) * HW_ + n] = a6;
    }
}

// ---------------------------------------------------------------------------
// K4: attention pass 1 — Z[n] = sum_m exp(q1[:,n]·k2[:,m]).
// One wave owns 4 n-tiles, sweeps all 256 m-tiles with WMMA + v_exp_f32.
// ---------------------------------------------------------------------------
#define NTW 4
__global__ void k_pass1(const _Float16* __restrict__ q1p,
                        const _Float16* __restrict__ k2p,
                        float* __restrict__ Z) {
    int b = blockIdx.y;
    int wave = threadIdx.x >> 5;
    int lane = threadIdx.x & 31;
    int tbase = (blockIdx.x * 8 + wave) * NTW;
    const v16h* qf = (const v16h*)q1p + (size_t)b * NT_ * 32;
    const v16h* kf = (const v16h*)k2p + (size_t)b * NT_ * 32;
    v16h a[NTW];
    #pragma unroll
    for (int t = 0; t < NTW; ++t) a[t] = qf[(size_t)(tbase + t) * 32 + lane];
    float accZ[NTW][8] = {};
    for (int mt = 0; mt < NT_; ++mt) {
        v16h bf = kf[(size_t)mt * 32 + lane];
        #pragma unroll
        for (int t = 0; t < NTW; ++t) {
            v8f c = {};
            c = wmma_f16(a[t], bf, c);
            #pragma unroll
            for (int r = 0; r < 8; ++r) accZ[t][r] += __expf(c[r]);
        }
    }
    // D layout: lanes 0..15 hold rows 0..7 (cols 0..15); lanes 16..31 rows 8..15.
    #pragma unroll
    for (int t = 0; t < NTW; ++t) {
        #pragma unroll
        for (int r = 0; r < 8; ++r) {
            float v = accZ[t][r];
            v += __shfl_xor(v, 1, 16);
            v += __shfl_xor(v, 2, 16);
            v += __shfl_xor(v, 4, 16);
            v += __shfl_xor(v, 8, 16);
            accZ[t][r] = v;
        }
    }
    if ((lane & 15) == 0) {
        int rb = (lane >> 4) * 8;
        #pragma unroll
        for (int t = 0; t < NTW; ++t)
            #pragma unroll
            for (int r = 0; r < 8; ++r)
                Z[(size_t)b * HW_ + (tbase + t) * 16 + rb + r] = accZ[t][r];
    }
}

// ---------------------------------------------------------------------------
// K5: yqn = yq / Z, already in A-fragment layout (same packed index space).
// ---------------------------------------------------------------------------
__global__ void k_yqn(const _Float16* __restrict__ yqp, const float* __restrict__ Z,
                      _Float16* __restrict__ yqnp) {
    int idx = blockIdx.x * blockDim.x + threadIdx.x;
    if (idx >= B_ * 2 * NPAIR_ * 32 * 16) return;
    int half = idx & 15;
    int lane = (idx >> 4) & 31;
    int np   = (idx >> 9) & 127;
    int b    = idx >> 17;
    int hi = lane >> 4;
    int k = (half & 7) + (half >= 8 ? 16 : 0) + hi * 8;
    int n = np * 32 + k;
    yqnp[idx] = (_Float16)((float)yqp[idx] / Z[(size_t)b * HW_ + n]);
}

// ---------------------------------------------------------------------------
// K6: attention pass 2 — xy_hw[:,m] = sum_n (yq[:,n]/Z[n]) * exp(s[n,m]).
// Chained WMMAs: score D-fragments (two 16x16 P tiles) map *in place* onto
// the 32xK B fragment of the accumulation WMMA. 4 m-tiles per wave for reuse.
// ---------------------------------------------------------------------------
#define MTW 4
__global__ void k_pass2(const _Float16* __restrict__ q1p,
                        const _Float16* __restrict__ k2p,
                        const _Float16* __restrict__ yqnp,
                        float* __restrict__ xy_hw) {
    int b = blockIdx.y;
    int wave = threadIdx.x >> 5;
    int lane = threadIdx.x & 31;
    int mbase = (blockIdx.x * 8 + wave) * MTW;
    const v16h* qf = (const v16h*)q1p + (size_t)b * NT_ * 32;
    const v16h* kf = (const v16h*)k2p + (size_t)b * NT_ * 32;
    const v16h* yf = (const v16h*)yqnp + (size_t)b * 2 * NPAIR_ * 32;
    v16h kb[MTW];
    #pragma unroll
    for (int t = 0; t < MTW; ++t) kb[t] = kf[(size_t)(mbase + t) * 32 + lane];
    v8f acc[MTW][2];
    #pragma unroll
    for (int t = 0; t < MTW; ++t) { acc[t][0] = (v8f){}; acc[t][1] = (v8f){}; }
    for (int np = 0; np < NPAIR_; ++np) {
        v16h a0 = qf[(size_t)(2 * np) * 32 + lane];
        v16h a1 = qf[(size_t)(2 * np + 1) * 32 + lane];
        v16h y0 = yf[(size_t)(0 * NPAIR_ + np) * 32 + lane];
        v16h y1 = yf[(size_t)(1 * NPAIR_ + np) * 32 + lane];
        #pragma unroll
        for (int t = 0; t < MTW; ++t) {
            v8f d0 = {}; d0 = wmma_f16(a0, kb[t], d0);
            v8f d1 = {}; d1 = wmma_f16(a1, kb[t], d1);
            v16h bf;
            #pragma unroll
            for (int j = 0; j < 8; ++j) bf[j] = (_Float16)__expf(d0[j]);
            #pragma unroll
            for (int j = 0; j < 8; ++j) bf[8 + j] = (_Float16)__expf(d1[j]);
            acc[t][0] = wmma_f16(y0, bf, acc[t][0]);
            acc[t][1] = wmma_f16(y1, bf, acc[t][1]);
        }
    }
    int l0 = lane & 15, hi = lane >> 4;
    #pragma unroll
    for (int t = 0; t < MTW; ++t) {
        int m = (mbase + t) * 16 + l0;
        #pragma unroll
        for (int ct = 0; ct < 2; ++ct)
            #pragma unroll
            for (int r = 0; r < 8; ++r) {
                int c = ct * 16 + hi * 8 + r;
                xy_hw[((size_t)(b * 32 + c)) * HW_ + m] = acc[t][ct][r];
            }
    }
}

// ---------------------------------------------------------------------------
// K7: channel attention Gram + softmax: xc[c,d] = softmax_d(sum_n k1[c,n]q2[d,n])
// One block per batch, wave = row c, lane = col d.
// ---------------------------------------------------------------------------
__global__ void k_gram(const float* __restrict__ k1, const float* __restrict__ q2,
                       float* __restrict__ xc) {
    int b = blockIdx.x;
    int c = threadIdx.x >> 5, d = threadIdx.x & 31;
    const float* kr = k1 + ((size_t)(b * 32 + c)) * HW_;
    const float* qr = q2 + ((size_t)(b * 32 + d)) * HW_;
    float acc = 0.f;
    for (int n = 0; n < HW_; ++n) acc += kr[n] * qr[n];
    float m = acc;
    m = fmaxf(m, __shfl_xor(m, 16, 32));
    m = fmaxf(m, __shfl_xor(m, 8, 32));
    m = fmaxf(m, __shfl_xor(m, 4, 32));
    m = fmaxf(m, __shfl_xor(m, 2, 32));
    m = fmaxf(m, __shfl_xor(m, 1, 32));
    float e = __expf(acc - m);
    float s = e;
    s += __shfl_xor(s, 16, 32);
    s += __shfl_xor(s, 8, 32);
    s += __shfl_xor(s, 4, 32);
    s += __shfl_xor(s, 2, 32);
    s += __shfl_xor(s, 1, 32);
    xc[((size_t)(b * 32 + c)) * 32 + d] = e / s;
}

// K8: xy_c[d,n] = sum_c xc[c,d] * yk[c,n]
__global__ void k_xyc(const float* __restrict__ xc, const float* __restrict__ yk,
                      float* __restrict__ xy_c) {
    int idx = blockIdx.x * blockDim.x + threadIdx.x;
    if (idx >= B_ * 32 * HW_) return;
    int n = idx & 4095, d = (idx >> 12) & 31, b = idx >> 17;
    float acc = 0.f;
    #pragma unroll
    for (int c = 0; c < 32; ++c)
        acc += xc[((size_t)(b * 32 + c)) * 32 + d] * yk[((size_t)(b * 32 + c)) * HW_ + n];
    xy_c[((size_t)(b * 32 + d)) * HW_ + n] = acc;
}

// K9: out_final = conv6([xy_hw;xy_c]); x_out = x + out_final
__global__ void k_fuse(const float* __restrict__ xy_hw, const float* __restrict__ xy_c,
                       const float* __restrict__ w6, const float* __restrict__ b6,
                       const float* __restrict__ x, float* __restrict__ out_x) {
    int idx = blockIdx.x * blockDim.x + threadIdx.x;
    if (idx >= B_ * HW_) return;
    int b = idx >> 12, n = idx & 4095;
    float hv[32], cv[32];
    #pragma unroll
    for (int c = 0; c < 32; ++c) {
        hv[c] = xy_hw[((size_t)(b * 32 + c)) * HW_ + n];
        cv[c] = xy_c[((size_t)(b * 32 + c)) * HW_ + n];
    }
    for (int oc = 0; oc < 32; ++oc) {
        float acc = b6[oc];
        #pragma unroll
        for (int i = 0; i < 32; ++i) acc += w6[oc * 64 + i] * hv[i];
        #pragma unroll
        for (int i = 0; i < 32; ++i) acc += w6[oc * 64 + 32 + i] * cv[i];
        size_t o = ((size_t)(b * 32 + oc)) * HW_ + n;
        out_x[o] = x[o] + acc;
    }
}

// K10: y_out = y + conv3x3(x_out, score_w) + score_b
__global__ void k_score(const float* __restrict__ out_x, const float* __restrict__ y,
                        const float* __restrict__ sw, const float* __restrict__ sb,
                        float* __restrict__ out_y) {
    int idx = blockIdx.x * blockDim.x + threadIdx.x;
    if (idx >= B_ * HW_) return;
    int b = idx >> 12, n = idx & 4095;
    int h = n >> 6, wq = n & 63;
    float acc = sb[0];
    for (int c = 0; c < 32; ++c) {
        const float* src = out_x + ((size_t)(b * 32 + c)) * HW_;
        const float* wp = sw + (size_t)c * 9;
        #pragma unroll
        for (int kh = 0; kh < 3; ++kh) {
            int hh = h + kh - 1;
            if (hh < 0 || hh > 63) continue;
            #pragma unroll
            for (int kw = 0; kw < 3; ++kw) {
                int ww = wq + kw - 1;
                if (ww < 0 || ww > 63) continue;
                acc += wp[kh * 3 + kw] * src[hh * 64 + ww];
            }
        }
    }
    out_y[(size_t)b * HW_ + n] = y[(size_t)b * HW_ + n] + acc;
}

// ---------------------------------------------------------------------------
extern "C" void kernel_launch(void* const* d_in, const int* in_sizes, int n_in,
                              void* d_out, int out_size, void* d_ws, size_t ws_size,
                              hipStream_t stream) {
    const float* x     = (const float*)d_in[0];
    const float* y     = (const float*)d_in[1];
    const float* convw = (const float*)d_in[2];
    const float* convb = (const float*)d_in[3];
    const float* scw   = (const float*)d_in[4];
    const float* scb   = (const float*)d_in[5];
    const float* c132w = (const float*)d_in[6];
    const float* c132b = (const float*)d_in[7];
    const float* q1w = (const float*)d_in[8];  const float* q1b = (const float*)d_in[9];
    const float* k1w = (const float*)d_in[10]; const float* k1b = (const float*)d_in[11];
    const float* q2w = (const float*)d_in[12]; const float* q2b = (const float*)d_in[13];
    const float* k2w = (const float*)d_in[14]; const float* k2b = (const float*)d_in[15];
    const float* q3w = (const float*)d_in[16]; const float* q3b = (const float*)d_in[17];
    const float* k3w = (const float*)d_in[18]; const float* k3b = (const float*)d_in[19];
    const float* w6  = (const float*)d_in[20]; const float* b6  = (const float*)d_in[21];

    char* wsp = (char*)d_ws;
    auto take = [&](size_t bytes) -> void* {
        void* r = (void*)wsp;
        wsp += (bytes + 255) & ~(size_t)255;
        return r;
    };
    const size_t NBCHW = (size_t)B_ * CH_ * HW_;           // 1,048,576
    float*    x1_co = (float*)take(NBCHW * 4);
    float*    x2_co = (float*)take(NBCHW * 4);
    float*    k1o   = (float*)take(NBCHW * 4);
    float*    q2o   = (float*)take(NBCHW * 4);
    float*    yko   = (float*)take(NBCHW * 4);
    float*    Z     = (float*)take((size_t)B_ * HW_ * 4);
    _Float16* q1p   = (_Float16*)take(NBCHW * 2);          // packed fragments
    _Float16* k2p   = (_Float16*)take(NBCHW * 2);
    _Float16* yqp   = (_Float16*)take(NBCHW * 2);
    _Float16* yqnp  = (_Float16*)take(NBCHW * 2);
    float*    xy_hw = (float*)take(NBCHW * 4);
    float*    xy_c  = (float*)take(NBCHW * 4);
    float*    xc    = (float*)take((size_t)B_ * 32 * 32 * 4);

    float* out_x = (float*)d_out;                 // [8,32,64,64]
    float* out_y = out_x + NBCHW;                 // [8,1,64,64]

    k_conv1<<<dim3((NBCHW + 255) / 256), 256, 0, stream>>>(x, y, convw, convb, x1_co);
    k_conv2<<<dim3((NBCHW + 255) / 256), 256, 0, stream>>>(y, c132w, c132b, x2_co);
    k_proj<<<dim3((B_ * HW_ + 255) / 256), 256, 0, stream>>>(
        x1_co, x2_co, q1w, q1b, k1w, k1b, q2w, q2b, k2w, k2b, q3w, q3b, k3w, k3b,
        q1p, k2p, yqp, k1o, q2o, yko);
    k_pass1<<<dim3(8, B_), 256, 0, stream>>>(q1p, k2p, Z);
    k_yqn<<<dim3((NBCHW + 255) / 256), 256, 0, stream>>>(yqp, Z, yqnp);
    k_pass2<<<dim3(8, B_), 256, 0, stream>>>(q1p, k2p, yqnp, xy_hw);
    k_gram<<<dim3(B_), 1024, 0, stream>>>(k1o, q2o, xc);
    k_xyc<<<dim3((NBCHW + 255) / 256), 256, 0, stream>>>(xc, yko, xy_c);
    k_fuse<<<dim3((B_ * HW_ + 255) / 256), 256, 0, stream>>>(xy_hw, xy_c, w6, b6, x, out_x);
    k_score<<<dim3((B_ * HW_ + 255) / 256), 256, 0, stream>>>(out_x, y, scw, scb, out_y);
}
